// MahalanobisDistanceLoss_10582799417965
// MI455X (gfx1250) — matrix operand order repaired and checked by reading, not compiled, not run on hardware
//
#include <hip/hip_runtime.h>

typedef __attribute__((ext_vector_type(16))) _Float16 v16h;
typedef __attribute__((ext_vector_type(2)))  _Float16 v2h;
typedef __attribute__((ext_vector_type(2)))  __fp16   v2hf;   // cvt_pkrtz native type
typedef __attribute__((ext_vector_type(8)))  float    v8f;

#define DIM 128
#define ROWSTRIDE 132   // padded diff row stride (floats); keeps LDS gathers near conflict-free
#define WAVES 8

// ---------------------------------------------------------------------------
// Kernel 1: Gauss-Jordan inversion of the 128x128 SPD covariance (1 block),
// then emit C^-1 in WMMA-B fragment-ready order, split into f16 hi/lo:
//   bfrag[ ((ct*4 + kc)*2 + s)*32 + lane ][ h ]   (16 contiguous halves/lane)
// where the lane's B 32x16 layout is K = kc*32 + h + (lane>=16 ? 16 : 0),
//       N = ct*16 + (lane & 15);   s = 0 -> hi, s = 1 -> lo residual.
// ---------------------------------------------------------------------------
__global__ __launch_bounds__(128) void invert_pack(const float* __restrict__ cov,
                                                   unsigned short* __restrict__ bfrag) {
    extern __shared__ float M[];        // 128 x 256 augmented matrix (128 KB)
    const int r = threadIdx.x;
    for (int j = 0; j < DIM; ++j) {
        M[r * 256 + j]       = cov[r * DIM + j];
        M[r * 256 + DIM + j] = (j == r) ? 1.0f : 0.0f;
    }
    __syncthreads();
    for (int k = 0; k < DIM; ++k) {
        if (r == k) {
            const float inv = 1.0f / M[k * 256 + k];
            for (int j = 0; j < 256; ++j) M[k * 256 + j] *= inv;
        }
        __syncthreads();
        if (r != k) {
            const float f = M[r * 256 + k];
            for (int j = 0; j < 256; ++j) M[r * 256 + j] -= f * M[k * 256 + j];
        }
        __syncthreads();
    }
    // Scatter into fragment-ready layout (32768 halves = 64 KB, written once)
    for (int idx = r; idx < 32768; idx += 128) {
        const int h    = idx & 15;
        const int lane = (idx >> 4) & 31;
        const int s    = (idx >> 9) & 1;
        const int kc   = (idx >> 10) & 3;
        const int ct   = (idx >> 12) & 7;
        const int K    = kc * 32 + h + ((lane >= 16) ? 16 : 0);
        const int N    = ct * 16 + (lane & 15);
        const float v  = M[K * 256 + DIM + N];
        const _Float16 hi = (_Float16)v;
        const _Float16 va = s ? (_Float16)(v - (float)hi) : hi;
        union { _Float16 f; unsigned short u; } cv; cv.f = va;
        bfrag[idx] = cv.u;
    }
}

// ---------------------------------------------------------------------------
// Kernel 2: per-row quadratic form via v_wmma_f32_16x16x32_f16 with hi/lo
// error compensation:  Y ~= Dh*Ch + Dh*Cl + Dl*Ch  (f32 accumulate).
// One wave handles a 16-row slab; block = 8 waves = 128 rows.
// ---------------------------------------------------------------------------
__global__ __launch_bounds__(256) void mahal_main(const float* __restrict__ e1,
                                                  const float* __restrict__ e2,
                                                  const unsigned short* __restrict__ bfrag,
                                                  const int* __restrict__ target,
                                                  float* __restrict__ partials) {
    extern __shared__ unsigned char smem[];
    _Float16* cS = (_Float16*)smem;                     // fragment-ready C^-1 (64 KB)
    float*    dS = (float*)(smem + 65536);              // 8 waves * 16 * ROWSTRIDE f32

    const int tid  = threadIdx.x;
    const int lane = tid & 31;
    const int wave = tid >> 5;
    const int blk  = blockIdx.x;

    // Stage fragment-ready C^-1 into LDS (block-wide, linear, vectorized)
    {
        const uint4* src = (const uint4*)bfrag;
        uint4*       dst = (uint4*)cS;
        #pragma unroll
        for (int i = 0; i < 4096 / 256; ++i)
            dst[tid + i * 256] = src[tid + i * 256];
    }

    // Stage this wave's 16x128 diff slab into LDS (padded row stride)
    {
        const int r0 = blk * (16 * WAVES) + wave * 16;
        const float4* p1 = (const float4*)(e1 + (size_t)r0 * DIM);
        const float4* p2 = (const float4*)(e2 + (size_t)r0 * DIM);
        float* d = dS + wave * (16 * ROWSTRIDE);
        #pragma unroll
        for (int i = 0; i < 16; ++i) {
            const int idx = lane + i * 32;          // 512 float4 per slab
            const int row = idx >> 5;
            const int c4  = idx & 31;
            const float4 a = p1[idx];
            const float4 b = p2[idx];
            float* o = d + row * ROWSTRIDE + c4 * 4;
            o[0] = a.x - b.x; o[1] = a.y - b.y; o[2] = a.z - b.z; o[3] = a.w - b.w;
        }
    }
    __syncthreads();

    // Build A fragments (16-bit A 16x32 layout). Every half-pair maps to
    // adjacent K values -> one ds_load_b64 + one v_cvt_pk per u32 lane.
    const float* drow   = dS + wave * (16 * ROWSTRIDE) + (lane & 15) * ROWSTRIDE;
    const int    kbaseA = (lane >= 16) ? 16 : 0;
    union AF { v16h v; v2h p[8]; };
    AF a_hi[4], a_lo[4];
    #pragma unroll
    for (int kc = 0; kc < 4; ++kc) {
        #pragma unroll
        for (int j = 0; j < 8; ++j) {
            const int Kb = kc * 32 + 2 * j + kbaseA;
            const float2 pv = *(const float2*)(drow + Kb);
            const v2hf hi2 = __builtin_amdgcn_cvt_pkrtz(pv.x, pv.y);
            const float r0 = pv.x - (float)hi2.x;
            const float r1 = pv.y - (float)hi2.y;
            a_hi[kc].p[j] = __builtin_bit_cast(v2h, hi2);
            a_lo[kc].p[j] = __builtin_bit_cast(v2h, __builtin_amdgcn_cvt_pkrtz(r0, r1));
        }
    }

    float qpart[8];
    #pragma unroll
    for (int v = 0; v < 8; ++v) qpart[v] = 0.0f;

    const int nB    = lane & 15;
    const int rbase = (lane >= 16) ? 8 : 0;     // C/D layout: lanes 16-31 hold M=v+8
    const v16h* bfr = (const v16h*)cS;          // 32 v16h per fragment (one per lane)

    #pragma unroll
    for (int ct = 0; ct < 8; ++ct) {
        v8f accA = {0.f,0.f,0.f,0.f,0.f,0.f,0.f,0.f};
        v8f accB = {0.f,0.f,0.f,0.f,0.f,0.f,0.f,0.f};
        v8f accC = {0.f,0.f,0.f,0.f,0.f,0.f,0.f,0.f};
        #pragma unroll
        for (int kc = 0; kc < 4; ++kc) {
            const v16h bh = bfr[(ct * 4 + kc) * 64 + lane];        // hi fragment
            const v16h bl = bfr[(ct * 4 + kc) * 64 + 32 + lane];   // lo fragment
            accA = __builtin_amdgcn_wmma_f32_16x16x32_f16(false, a_hi[kc].v, false, bh,
                                                          (short)0, accA, false, false);
            accB = __builtin_amdgcn_wmma_f32_16x16x32_f16(false, a_hi[kc].v, false, bl,
                                                          (short)0, accB, false, false);
            accC = __builtin_amdgcn_wmma_f32_16x16x32_f16(false, a_lo[kc].v, false, bh,
                                                          (short)0, accC, false, false);
        }
        // quad partial: dot Y tile against diff using the C/D VGPR layout
        const float* dcol = dS + wave * (16 * ROWSTRIDE) + ct * 16 + nB;
        #pragma unroll
        for (int v = 0; v < 8; ++v) {
            const float y = accA[v] + accB[v] + accC[v];
            qpart[v] += y * dcol[(v + rbase) * ROWSTRIDE];
        }
    }

    // Reduce across the 16-lane column groups
    #pragma unroll
    for (int v = 0; v < 8; ++v) {
        float q = qpart[v];
        q += __shfl_xor(q, 1, 32);
        q += __shfl_xor(q, 2, 32);
        q += __shfl_xor(q, 4, 32);
        q += __shfl_xor(q, 8, 32);
        qpart[v] = q;   // every lane in its 16-group now holds the row sum
    }

    // Lane 0 -> rows r0..r0+7, lane 16 -> rows r0+8..r0+15
    float wsum = 0.0f;
    if ((lane & 15) == 0) {
        const int r0 = blk * (16 * WAVES) + wave * 16 + rbase;
        #pragma unroll
        for (int v = 0; v < 8; ++v) {
            const float d = sqrtf(fmaxf(qpart[v], 0.0f));
            const float t = (float)target[r0 + v];
            wsum += t * d + (1.0f - t) * fmaxf(0.0f, 1.0f - d);
        }
    }
    wsum += __shfl_xor(wsum, 16, 32);
    if (lane == 0) partials[blk * WAVES + wave] = wsum;
}

// ---------------------------------------------------------------------------
// Kernel 3: deterministic fixed-order reduction of per-wave partials -> mean
// ---------------------------------------------------------------------------
__global__ __launch_bounds__(256) void reduce_finalize(const float* __restrict__ partials,
                                                       float* __restrict__ out,
                                                       int n, float invB) {
    __shared__ float s[256];
    float sum = 0.0f;
    for (int i = threadIdx.x; i < n; i += 256) sum += partials[i];
    s[threadIdx.x] = sum;
    __syncthreads();
    for (int w = 128; w > 0; w >>= 1) {
        if (threadIdx.x < w) s[threadIdx.x] += s[threadIdx.x + w];
        __syncthreads();
    }
    if (threadIdx.x == 0) out[0] = s[0] * invB;
}

// ---------------------------------------------------------------------------
extern "C" void kernel_launch(void* const* d_in, const int* in_sizes, int n_in,
                              void* d_out, int out_size, void* d_ws, size_t ws_size,
                              hipStream_t stream) {
    const float* e1     = (const float*)d_in[0];
    const float* e2     = (const float*)d_in[1];
    const float* cov    = (const float*)d_in[2];
    const int*   target = (const int*)d_in[3];
    float*       out    = (float*)d_out;

    const long long B  = (long long)in_sizes[0] / DIM;   // 262144
    const int slabs    = (int)(B / (16 * WAVES));        // 128 rows per block

    unsigned short* bfrag = (unsigned short*)d_ws;                  // 64 KB
    float* partials = (float*)((char*)d_ws + 32768 * sizeof(unsigned short));

    const size_t lds_inv  = 128 * 256 * sizeof(float);                 // 128 KB
    const size_t lds_main = 65536 + WAVES * 16 * ROWSTRIDE * 4;        // 64K + 66K

    invert_pack<<<dim3(1), dim3(128), lds_inv, stream>>>(cov, bfrag);
    mahal_main<<<dim3(slabs), dim3(256), lds_main, stream>>>(e1, e2, bfrag, target, partials);
    reduce_finalize<<<dim3(1), dim3(256), 0, stream>>>(partials, out, slabs * WAVES,
                                                       1.0f / (float)B);
}
